// MixtralMoE_51634096832583
// MI455X (gfx1250) — matrix operand order repaired and checked by reading, not compiled, not run on hardware
//
#include <hip/hip_runtime.h>
#include <math.h>

// ---- problem constants (match reference) ----
#define T_TOK 8192
#define DIM   1024
#define FDIM  2048
#define NEXP  8
#define MT    32                 // tokens per block tile (2 x 16-row WMMA tiles)
#define TILES (T_TOK / MT)       // 256 tiles per expert (worst case)

typedef __attribute__((ext_vector_type(16))) __bf16 v16bf;
typedef __attribute__((ext_vector_type(8)))  float  v8f;
typedef unsigned short ushort_t;

__device__ __forceinline__ ushort_t f2bf(float x) {
  union { float f; unsigned u; } v; v.f = x;
  unsigned r = v.u + 0x7FFFu + ((v.u >> 16) & 1u);   // round-to-nearest-even
  return (ushort_t)(r >> 16);
}

__device__ __forceinline__ v8f zero8() {
  v8f z = {0.f, 0.f, 0.f, 0.f, 0.f, 0.f, 0.f, 0.f};
  return z;
}

// load a 32-byte WMMA fragment slice (16 bf16) as two 16B chunks
// (ds_load_b128 x2 from LDS, global_load_b128 x2 from global)
__device__ __forceinline__ v16bf ldfrag(const ushort_t* p) {
  union { uint4 q[2]; v16bf v; } u;
  u.q[0] = *reinterpret_cast<const uint4*>(p);
  u.q[1] = *reinterpret_cast<const uint4*>(p + 8);
  return u.v;
}

#define WMMA_BF16(A, B, C) \
  __builtin_amdgcn_wmma_f32_16x16x32_bf16(false, (A), false, (B), (short)0, (C), false, false)

// ---------------------------------------------------------------------------
// Convert fp32 weights -> bf16 AND pack into WMMA B-fragment order.
// B fragment (32K x 16N, bf16): lane = n + 16*((k>>3)&1), half = (k&7) + 8*(k>>4).
// Output layout: frag[((e*(N/16) + ft) * (K/32) + kt)] of 512 halfs,
// within frag: lane*16 + half  -> one frag = 1KB contiguous.
// Block handles a [32K x 128N] source tile; grid = E*(K/32)*(N/128).
// ---------------------------------------------------------------------------
__global__ void cvt_pack_b(const float* __restrict__ src, ushort_t* __restrict__ dst,
                           int K, int N) {
  __shared__ ushort_t sb[8 * 512];           // 8 fragments, 8KB
  int tilesPerExpert = (K / 32) * (N / 128);
  int e    = blockIdx.x / tilesPerExpert;
  int r    = blockIdx.x % tilesPerExpert;
  int kt   = r % (K / 32);
  int n128 = r / (K / 32);

  const float* s = src + (long)e * K * N + (long)(kt * 32) * N + n128 * 128;

  int tid = threadIdx.x;
  int rg = tid >> 5;            // 0..7  -> rows k0 = rg*4 .. +3
  int cg = tid & 31;            // 0..31 -> cols n0 = cg*4 .. +3
  int k0 = rg * 4;
  int n0 = cg * 4;

  float4 row[4];
#pragma unroll
  for (int i = 0; i < 4; ++i)
    row[i] = *reinterpret_cast<const float4*>(s + (long)(k0 + i) * N + n0);

#pragma unroll
  for (int j = 0; j < 4; ++j) {
    int n = n0 + j;
    float c0 = (j == 0) ? row[0].x : (j == 1) ? row[0].y : (j == 2) ? row[0].z : row[0].w;
    float c1 = (j == 0) ? row[1].x : (j == 1) ? row[1].y : (j == 2) ? row[1].z : row[1].w;
    float c2 = (j == 0) ? row[2].x : (j == 1) ? row[2].y : (j == 2) ? row[2].z : row[2].w;
    float c3 = (j == 0) ? row[3].x : (j == 1) ? row[3].y : (j == 2) ? row[3].z : row[3].w;
    uint2 p;
    p.x = (unsigned)f2bf(c0) | ((unsigned)f2bf(c1) << 16);
    p.y = (unsigned)f2bf(c2) | ((unsigned)f2bf(c3) << 16);
    int ft   = n >> 4;                         // 0..7 within this 128-col slab
    int lane = (n & 15) + 16 * ((k0 >> 3) & 1);
    int hb   = (k0 & 7) | (((k0 >> 4) & 1) << 3); // 0,4,8,12 -> 8B aligned
    *reinterpret_cast<uint2*>(&sb[ft * 512 + lane * 16 + hb]) = p;
  }
  __syncthreads();

  // coalesced write-out: each fragment is a contiguous 1KB block in dst
  int f   = tid >> 5;                          // fragment 0..7
  int off = (tid & 31) * 16;                   // 32B per thread
  long fragIdx = ((long)e * (N / 16) + (long)n128 * 8 + f) * (K / 32) + kt;
  const uint4* ls = reinterpret_cast<const uint4*>(&sb[f * 512 + off]);
  uint4* gd = reinterpret_cast<uint4*>(dst + fragIdx * 512 + off);
  gd[0] = ls[0];
  gd[1] = ls[1];
}

// ---------------- router: one wave32 per token ----------------
__global__ void router_kernel(const float* __restrict__ hidden,
                              const float* __restrict__ gate,
                              int*   __restrict__ counts,
                              int*   __restrict__ stok,
                              float* __restrict__ swgt) {
  int lane = threadIdx.x & 31;
  int wave = threadIdx.x >> 5;
  int t = blockIdx.x * 8 + wave;
  const float* hp = hidden + (long)t * DIM;

  float acc[NEXP];
#pragma unroll
  for (int e = 0; e < NEXP; ++e) acc[e] = 0.f;

  for (int i = 0; i < DIM / 32; ++i) {
    int d = lane + 32 * i;
    float x = hp[d];
    const float4* g = reinterpret_cast<const float4*>(gate + (long)d * NEXP);
    float4 g0 = g[0], g1 = g[1];
    acc[0] += x * g0.x; acc[1] += x * g0.y; acc[2] += x * g0.z; acc[3] += x * g0.w;
    acc[4] += x * g1.x; acc[5] += x * g1.y; acc[6] += x * g1.z; acc[7] += x * g1.w;
  }
#pragma unroll
  for (int off = 16; off > 0; off >>= 1) {
#pragma unroll
    for (int e = 0; e < NEXP; ++e) acc[e] += __shfl_xor(acc[e], off, 32);
  }
  if (lane == 0) {
    int i1 = 0; float v1 = acc[0];
#pragma unroll
    for (int e = 1; e < NEXP; ++e) if (acc[e] > v1) { v1 = acc[e]; i1 = e; }
    int i2 = -1; float v2 = -3.0e38f;
#pragma unroll
    for (int e = 0; e < NEXP; ++e) if (e != i1 && acc[e] > v2) { v2 = acc[e]; i2 = e; }
    // top-2 of softmax then renormalize == exp(l-max) pair normalized
    float e2 = __expf(v2 - v1);
    float inv = __builtin_amdgcn_rcpf(1.f + e2);
    float w1 = inv, w2 = e2 * inv;
    int p1 = atomicAdd(&counts[i1], 1);
    stok[i1 * T_TOK + p1] = t; swgt[i1 * T_TOK + p1] = w1;
    int p2 = atomicAdd(&counts[i2], 1);
    stok[i2 * T_TOK + p2] = t; swgt[i2 * T_TOK + p2] = w2;
  }
}

// ---------------------------------------------------------------------------
// Fused routed SwiGLU expert MLP. grid = NEXP*TILES blocks, 256 threads.
// Weights are pre-packed B-fragments -> loaded global->VGPR directly
// (per-lane 32B contiguous, whole wave = one 1KB fragment, L2-resident).
// No LDS staging / no barriers inside the K loops. 1 N-tile per wave,
// 4 accumulator chains -> fits registers with zero spills; the compiler
// unrolls 4x and clauses the global fragment loads for latency hiding.
// Dynamic LDS = 192 KB: X frags 64 KB + act frags 128 KB.
// ---------------------------------------------------------------------------
__global__ void __launch_bounds__(256)
moe_kernel(const float* __restrict__ hidden,
           const int*   __restrict__ counts,
           const int*   __restrict__ stokAll,
           const float* __restrict__ swgtAll,
           const ushort_t* __restrict__ w1f,
           const ushort_t* __restrict__ w3f,
           const ushort_t* __restrict__ w2f,
           float* __restrict__ out) {
  int e  = blockIdx.x >> 8;
  int mt = blockIdx.x & (TILES - 1);
  int cnt = counts[e];
  if (mt * MT >= cnt) return;                  // uniform early exit

  const int*   stok = stokAll + e * T_TOK + mt * MT;
  const float* swgt = swgtAll + e * T_TOK + mt * MT;

  extern __shared__ char smemRaw[];
  ushort_t* aFrag   = reinterpret_cast<ushort_t*>(smemRaw);
  ushort_t* actFrag = aFrag + 2 * 32 * 512;

  int tid  = threadIdx.x;
  int lane = tid & 31;
  int wave = tid >> 5;

  // ---- stage X tile: gather routed rows, fp32->bf16, A-fragment packed.
  // 8 consecutive (8-aligned) K values share a lane and consecutive halfs
  // -> one ds_store_b128 per chunk.
  {
    int m  = tid >> 3;              // 0..31 token row in tile
    int c0 = (tid & 7) * 128;       // column chunk
    bool valid = (mt * MT + m) < cnt;
    int token = valid ? stok[m] : 0;
    const float4* src = reinterpret_cast<const float4*>(hidden + (long)token * DIM + c0);
#pragma unroll 2
    for (int ch = 0; ch < 16; ++ch) {          // 16 chunks of 8 elements
      float4 v0 = src[ch * 2];
      float4 v1 = src[ch * 2 + 1];
      uint4 p;
      p.x = (unsigned)f2bf(v0.x) | ((unsigned)f2bf(v0.y) << 16);
      p.y = (unsigned)f2bf(v0.z) | ((unsigned)f2bf(v0.w) << 16);
      p.z = (unsigned)f2bf(v1.x) | ((unsigned)f2bf(v1.y) << 16);
      p.w = (unsigned)f2bf(v1.z) | ((unsigned)f2bf(v1.w) << 16);
      int k  = c0 + ch * 8;
      int kt = k >> 5, kp = k & 31;
      int ln = (m & 15) + 16 * ((kp >> 3) & 1);
      int hb = ((kp >> 4) & 1) << 3;           // 0 or 8 -> 16B aligned
      *reinterpret_cast<uint4*>(&aFrag[((m >> 4) * 32 + kt) * 512 + ln * 16 + hb]) = p;
    }
  }
  __syncthreads();

  const ushort_t* w1p = w1f + ((long)e * (FDIM / 16)) * (DIM / 32) * 512;
  const ushort_t* w3p = w3f + ((long)e * (FDIM / 16)) * (DIM / 32) * 512;
  const ushort_t* w2p = w2f + ((long)e * (DIM / 16)) * (FDIM / 32) * 512;

  // ---- GEMM1: h1 = X*W1, h3 = X*W3; act = silu(h1)*h3 -> LDS (A-layout) ----
  for (int fb = 0; fb < FDIM / 128; ++fb) {
    int ft = fb * 8 + wave;                    // this wave's 16-col F tile
    const ushort_t* b1p = w1p + ((long)ft * (DIM / 32)) * 512 + lane * 16;
    const ushort_t* b3p = w3p + ((long)ft * (DIM / 32)) * 512 + lane * 16;
    v8f acc1[2] = {zero8(), zero8()};
    v8f acc3[2] = {zero8(), zero8()};
#pragma unroll 4
    for (int kb = 0; kb < DIM / 32; ++kb) {
      v16bf b1 = ldfrag(b1p + kb * 512);
      v16bf b3 = ldfrag(b3p + kb * 512);
      v16bf a0 = ldfrag(aFrag + (0 * 32 + kb) * 512 + lane * 16);
      v16bf a1 = ldfrag(aFrag + (1 * 32 + kb) * 512 + lane * 16);
      acc1[0] = WMMA_BF16(a0, b1, acc1[0]);
      acc1[1] = WMMA_BF16(a1, b1, acc1[1]);
      acc3[0] = WMMA_BF16(a0, b3, acc3[0]);
      acc3[1] = WMMA_BF16(a1, b3, acc3[1]);
    }
    // SwiGLU epilogue: silu via fast v_rcp_f32; scatter into act A-fragments
#pragma unroll
    for (int mtile = 0; mtile < 2; ++mtile) {
#pragma unroll
      for (int r = 0; r < 8; ++r) {
        float h1 = acc1[mtile][r];
        float h3 = acc3[mtile][r];
        float a  = h1 * __builtin_amdgcn_rcpf(1.f + __expf(-h1)) * h3;
        int mrow = mtile * 16 + r + 8 * (lane >> 4);
        int f    = fb * 128 + wave * 16 + (lane & 15);
        int kp = f & 31, kt2 = f >> 5;
        int ln = (mrow & 15) + 16 * ((kp >> 3) & 1);
        int hf = (kp & 7) | (((kp >> 4) & 1) << 3);
        actFrag[((mrow >> 4) * 64 + kt2) * 512 + ln * 16 + hf] = f2bf(a);
      }
    }
  }
  __syncthreads();

  // ---- GEMM2: out_tile = act * W2; scale by routing weight; scatter-add ----
  for (int nb = 0; nb < DIM / 128; ++nb) {
    int nt = nb * 8 + wave;                    // this wave's 16-col D tile
    const ushort_t* b2p = w2p + ((long)nt * (FDIM / 32)) * 512 + lane * 16;
    v8f acc[2] = {zero8(), zero8()};
#pragma unroll 4
    for (int kb = 0; kb < FDIM / 32; ++kb) {
      v16bf b  = ldfrag(b2p + kb * 512);
      v16bf a0 = ldfrag(actFrag + (0 * 64 + kb) * 512 + lane * 16);
      v16bf a1 = ldfrag(actFrag + (1 * 64 + kb) * 512 + lane * 16);
      acc[0] = WMMA_BF16(a0, b, acc[0]);
      acc[1] = WMMA_BF16(a1, b, acc[1]);
    }
    int n = nb * 128 + wave * 16 + (lane & 15);
#pragma unroll
    for (int mtile = 0; mtile < 2; ++mtile) {
#pragma unroll
      for (int r = 0; r < 8; ++r) {
        int mrow = mtile * 16 + r + 8 * (lane >> 4);
        if (mt * MT + mrow < cnt) {
          float wgt = swgt[mrow];
          atomicAdd(out + (long)stok[mrow] * DIM + n, acc[mtile][r] * wgt);
        }
      }
    }
  }
}

// ---------------- launch ----------------
extern "C" void kernel_launch(void* const* d_in, const int* in_sizes, int n_in,
                              void* d_out, int out_size, void* d_ws, size_t ws_size,
                              hipStream_t stream) {
  const float* hidden = (const float*)d_in[0];
  const float* gate   = (const float*)d_in[1];
  const float* w1     = (const float*)d_in[2];
  const float* w3     = (const float*)d_in[3];
  const float* w2     = (const float*)d_in[4];
  float* out = (float*)d_out;

  // workspace layout (~96.5 MB)
  char* ws = (char*)d_ws;
  int*   counts = (int*)ws;                                    // 8 ints (256B region)
  int*   stok   = (int*)(ws + 256);                            // E*T ints   (256 KB)
  float* swgt   = (float*)(ws + 256 + NEXP * T_TOK * 4);       // E*T floats (256 KB)
  size_t wbOff  = 256 + (size_t)NEXP * T_TOK * 8;              // 256B aligned
  const size_t WELTS = (size_t)NEXP * DIM * FDIM;              // 16,777,216 per tensor
  ushort_t* w1f = (ushort_t*)(ws + wbOff);
  ushort_t* w3f = w1f + WELTS;
  ushort_t* w2f = w3f + WELTS;

  hipMemsetAsync(d_ws, 0, 256, stream);                        // zero counts
  hipMemsetAsync(d_out, 0, (size_t)T_TOK * DIM * sizeof(float), stream);

  // convert + pack weights into B-fragment order (once per launch, L2-resident after)
  int packBlocks = NEXP * (DIM / 32) * (FDIM / 128);           // 4096 (w1/w3)
  cvt_pack_b<<<packBlocks, 256, 0, stream>>>(w1, w1f, DIM, FDIM);
  cvt_pack_b<<<packBlocks, 256, 0, stream>>>(w3, w3f, DIM, FDIM);
  int packBlocks2 = NEXP * (FDIM / 32) * (DIM / 128);          // 4096 (w2)
  cvt_pack_b<<<packBlocks2, 256, 0, stream>>>(w2, w2f, FDIM, DIM);

  router_kernel<<<T_TOK / 8, 256, 0, stream>>>(hidden, gate, counts, stok, swgt);

  size_t ldsBytes = (size_t)(2 * 32 * 512 + 2 * 64 * 512) * sizeof(ushort_t); // 192 KB
  moe_kernel<<<NEXP * TILES, 256, ldsBytes, stream>>>(hidden, counts, stok, swgt,
                                                      w1f, w3f, w2f, out);
}
